// Attention_59064390255178
// MI455X (gfx1250) — compile-verified, matrix operand order checked by reading, not run on hardware
//
#include <hip/hip_runtime.h>
#include <hip/hip_bf16.h>

// Problem dims (fixed by reference)
#define BB 4
#define NN 2048
#define CC 1024
#define HH 16
#define DD 64

typedef __attribute__((ext_vector_type(16))) __bf16 v16bf;
typedef __attribute__((ext_vector_type(8)))  __bf16 v8bf;
typedef __attribute__((ext_vector_type(8)))  float  v8f;

union V16U { v16bf v; v8bf h[2]; };

#if defined(__HIP_DEVICE_COMPILE__) && __has_builtin(__builtin_amdgcn_tensor_load_to_lds)
#define USE_TDM 1
#else
#define USE_TDM 0
#endif

__device__ __forceinline__ unsigned short f2bf(float f) {
  unsigned int u = __float_as_uint(f);
  u += 0x7FFFu + ((u >> 16) & 1u);   // round-to-nearest-even
  return (unsigned short)(u >> 16);
}

__device__ __forceinline__ v16bf load_frag(const unsigned short* p, int e0, int e1) {
  V16U u;
  u.h[0] = *reinterpret_cast<const v8bf*>(p + e0);
  u.h[1] = *reinterpret_cast<const v8bf*>(p + e1);
  return u.v;
}

__device__ __forceinline__ v8f wmma_bf16(v16bf a, v16bf b, v8f c) {
  // D = A(16x32 bf16) * B(32x16 bf16) + C(16x16 f32)
  return __builtin_amdgcn_wmma_f32_16x16x32_bf16(false, a, false, b, (short)0, c,
                                                 false, false);
}

__device__ __forceinline__ v8f zero8() {
  v8f z;
#pragma unroll
  for (int i = 0; i < 8; ++i) z[i] = 0.0f;
  return z;
}

#if USE_TDM
typedef __attribute__((ext_vector_type(4))) unsigned int v4u_t;
typedef __attribute__((ext_vector_type(8))) int          v8i_t;
typedef __attribute__((ext_vector_type(4))) int          v4i_t;

// Build a 2D Tensor-DMA descriptor (D#) per CDNA5 ISA and
// issue TENSOR_LOAD_TO_LDS. data_size = 2 bytes (bf16). Uniform (SGPR) operands.
// This toolchain's builtin takes 6 args (g0, g1, g2, g3, extra group, cpol).
//   lds_byte_off : byte offset of tile start within this workgroup's LDS
//   gptr         : global address of tile start
//   tile_d0/d1   : tile size, d0 = contiguous dimension (elements)
//   tensor_d0/d1 : tensor extents for OOB clamping (elements)
//   stride0      : elements between consecutive dim-1 lines
__device__ __forceinline__ void tdm_load_tile_2d(unsigned lds_byte_off,
                                                 const unsigned short* gptr,
                                                 unsigned tile_d0, unsigned tile_d1,
                                                 unsigned tensor_d0, unsigned tensor_d1,
                                                 unsigned long long stride0) {
  unsigned long long ga = (unsigned long long)(uintptr_t)gptr;
  v4u_t g0;
  g0[0] = 1u;                                        // count=1, user descriptor
  g0[1] = lds_byte_off;                              // lds_addr (bytes)
  g0[2] = (unsigned)(ga & 0xFFFFFFFFu);              // global_addr[31:0]
  g0[3] = (unsigned)((ga >> 32) & 0x01FFFFFFu)       // global_addr[56:32]
        | (2u << 30);                                // type = 2 ("image")
  v8i_t g1;
  g1[0] = (int)(1u << 16);                           // wg_mask=0, data_size=1 (2B)
  g1[1] = (int)((tensor_d0 & 0xFFFFu) << 16);        // tensor_dim0[15:0] @63:48
  g1[2] = (int)(((tensor_d0 >> 16) & 0xFFFFu)        // tensor_dim0[31:16]
        | ((tensor_d1 & 0xFFFFu) << 16));            // tensor_dim1[15:0]
  g1[3] = (int)(((tensor_d1 >> 16) & 0xFFFFu)        // tensor_dim1[31:16]
        | ((tile_d0 & 0xFFFFu) << 16));              // tile_dim0
  g1[4] = (int)(tile_d1 & 0xFFFFu);                  // tile_dim1 (tile_dim2=0)
  g1[5] = (int)(stride0 & 0xFFFFFFFFull);            // tensor_dim0_stride[31:0]
  g1[6] = (int)((stride0 >> 32) & 0xFFFFull);        // stride[47:32], dim1_stride lo=0
  g1[7] = 0;
  v4i_t gz4 = {0, 0, 0, 0};
  v8i_t gz8 = {0, 0, 0, 0, 0, 0, 0, 0};
  __builtin_amdgcn_tensor_load_to_lds(g0, g1, gz4, gz4, gz8, 0);
}
#endif

// ---------------------------------------------------------------------------
// fp32 -> bf16 cast (grid-stride)
// ---------------------------------------------------------------------------
__global__ void cast_bf16_kernel(const float* __restrict__ in,
                                 unsigned short* __restrict__ out, int n) {
  int i = blockIdx.x * blockDim.x + threadIdx.x;
  int stride = gridDim.x * blockDim.x;
  for (; i < n; i += stride) out[i] = f2bf(in[i]);
}

// ---------------------------------------------------------------------------
// QKV projection + fused RMSNorm + RoPE.
// One wave per 32(M) x 64(N) tile (B fragments reused across two A rows).
//   hb   : [B*N][C]  bf16 activations
//   wq   : [3C][C]   bf16 weights (rows = outputs -> natural B layout)
//   qb,kb: [B*H][N][64]  bf16
//   vb   : [B*H][64][N]  bf16 (transposed for P*V B-fragments)
// ---------------------------------------------------------------------------
__global__ __launch_bounds__(32) void qkv_rope_kernel(
    const unsigned short* __restrict__ hb, const unsigned short* __restrict__ wq,
    const float* __restrict__ cosb, const float* __restrict__ sinb,
    const float* __restrict__ nqw, const float* __restrict__ nkw,
    unsigned short* __restrict__ qb, unsigned short* __restrict__ kb,
    unsigned short* __restrict__ vb) {
  const int lane = threadIdx.x & 31;
  const int col  = lane & 15;
  const int kh   = lane >> 4;   // K-half select per ISA A/B layout

  const int tile  = blockIdx.x;        // 256 m-tiles * 48 n-tiles
  const int mTile = tile & 255;
  const int nTile = tile >> 8;
  const int mBase = mTile * 32;
  const int oBase = nTile * 64;
  const int sel   = oBase >> 10;           // 0=q, 1=k, 2=v
  const int head  = (oBase & 1023) >> 6;

  v8f acc[2][4];
#pragma unroll
  for (int r = 0; r < 2; ++r)
#pragma unroll
    for (int t = 0; t < 4; ++t) acc[r][t] = zero8();

  const unsigned short* arow0 = hb + (size_t)(mBase + col) * CC;
  const unsigned short* arow1 = hb + (size_t)(mBase + 16 + col) * CC;
  for (int k0 = 0; k0 < CC; k0 += 32) {
    v16bf a0 = load_frag(arow0, k0 + kh * 8, k0 + 16 + kh * 8);
    v16bf a1 = load_frag(arow1, k0 + kh * 8, k0 + 16 + kh * 8);
#pragma unroll
    for (int t = 0; t < 4; ++t) {
      const unsigned short* brow = wq + (size_t)(oBase + t * 16 + col) * CC;
      v16bf b = load_frag(brow, k0 + kh * 8, k0 + 16 + kh * 8);
      acc[0][t] = wmma_bf16(a0, b, acc[0][t]);
      acc[1][t] = wmma_bf16(a1, b, acc[1][t]);
    }
  }

  const int bIdx  = mBase / NN;
  const int nBase = mBase % NN;
  const size_t bh = (size_t)(bIdx * HH + head);

  if (sel == 2) {  // V: store transposed [bh][d][n]
    const size_t base = bh * (size_t)DD * NN;
#pragma unroll
    for (int r = 0; r < 2; ++r)
#pragma unroll
      for (int t = 0; t < 4; ++t)
#pragma unroll
        for (int j = 0; j < 8; ++j) {
          int row = r * 16 + j + 8 * kh;
          int d   = t * 16 + col;
          vb[base + (size_t)d * NN + nBase + row] = f2bf(acc[r][t][j]);
        }
    return;
  }

  const float* nw = (sel == 0) ? nqw : nkw;
  unsigned short* dst = (sel == 0) ? qb : kb;
  const size_t obase = bh * (size_t)NN * DD;

#pragma unroll
  for (int r = 0; r < 2; ++r) {
    // RMS norm over D=64: rows live in VGPR index j (j / j+8 per lane-half),
    // 64 elements of a row spread over 4 tiles x 16 lanes.
    float rinv[8];
#pragma unroll
    for (int j = 0; j < 8; ++j) {
      float v2 = 0.f;
#pragma unroll
      for (int t = 0; t < 4; ++t) { float x = acc[r][t][j]; v2 += x * x; }
      v2 += __shfl_xor(v2, 1, 32);
      v2 += __shfl_xor(v2, 2, 32);
      v2 += __shfl_xor(v2, 4, 32);
      v2 += __shfl_xor(v2, 8, 32);
      rinv[j] = rsqrtf(v2 * (1.0f / 64.0f) + 1e-6f);
    }

    float nv[4][8];
#pragma unroll
    for (int t = 0; t < 4; ++t) {
      float w = nw[t * 16 + col];
#pragma unroll
      for (int j = 0; j < 8; ++j) nv[t][j] = acc[r][t][j] * rinv[j] * w;
    }

    // RoPE: partner element at d +/- 32 sits in accumulator tile t +/- 2.
#pragma unroll
    for (int t = 0; t < 4; ++t)
#pragma unroll
      for (int j = 0; j < 8; ++j) {
        int row = r * 16 + j + 8 * kh;
        int n   = nBase + row;
        int d   = t * 16 + col;
        float c = cosb[(size_t)n * DD + d];
        float s = sinb[(size_t)n * DD + d];
        float rot = (t < 2) ? -nv[t + 2][j] : nv[t - 2][j];
        dst[obase + (size_t)n * DD + d] = f2bf(nv[t][j] * c + rot * s);
      }
  }
}

// ---------------------------------------------------------------------------
// Flash attention: one wave per (b,h, 16 q-tokens). Online softmax in fp32,
// scores and P*V through bf16 WMMA. K/V tiles staged into LDS by the Tensor
// Data Mover (double-buffered, TENSORcnt-tracked) when available.
// ---------------------------------------------------------------------------
// LDS element offsets (ushort units); static shared block starts at LDS offset 0.
#define KBUF_OFF 0
#define VBUF_OFF (2 * 64 * 64)
#define PBUF_OFF (4 * 64 * 64)
#define SMEM_ELTS (4 * 64 * 64 + 16 * 64)

__global__ __launch_bounds__(32) void flash_attn_kernel(
    const unsigned short* __restrict__ qb, const unsigned short* __restrict__ kb,
    const unsigned short* __restrict__ vb, unsigned short* __restrict__ attn) {
  __shared__ unsigned short smem[SMEM_ELTS];

  const int lane = threadIdx.x & 31;
  const int col  = lane & 15;
  const int kh   = lane >> 4;

  const int blk   = blockIdx.x;      // B*H*(N/16) = 8192
  const int qTile = blk & 127;
  const int bh    = blk >> 7;
  const int b     = bh >> 4;
  const int head  = bh & 15;

  const unsigned short* qp = qb + ((size_t)bh * NN + qTile * 16) * DD;
  const unsigned short* kp = kb + (size_t)bh * NN * DD;   // [n][d]
  const unsigned short* vp = vb + (size_t)bh * DD * NN;   // [d][n]

  // q fragments, resident for whole K loop (A layout: row=col, K-half=kh)
  v16bf qa0 = load_frag(qp + (size_t)col * DD, kh * 8, 16 + kh * 8);
  v16bf qa1 = load_frag(qp + (size_t)col * DD, 32 + kh * 8, 48 + kh * 8);

  float mrow[8], srow[8];
#pragma unroll
  for (int j = 0; j < 8; ++j) { mrow[j] = -3.0e38f; srow[j] = 0.f; }
  v8f o[4];
#pragma unroll
  for (int t = 0; t < 4; ++t) o[t] = zero8();

  const float scale = 0.125f;  // D^-0.5

#if USE_TDM
  // Prefetch chunk 0: K tile 64 tokens x 64 d ([n][d], stride 64),
  //                   V tile 64 d x 64 tokens ([d][n], stride NN).
  tdm_load_tile_2d((KBUF_OFF + 0) * 2, kp, DD, 64, DD, NN, DD);
  tdm_load_tile_2d((VBUF_OFF + 0) * 2, vp, 64, DD, NN, DD, NN);
#endif

  for (int kc = 0; kc < NN; kc += 64) {
#if USE_TDM
    const int par = (kc >> 6) & 1;
    if (kc + 64 < NN) {  // prefetch next chunk into other buffer
      const int nxt = par ^ 1;
      tdm_load_tile_2d((KBUF_OFF + nxt * 64 * 64) * 2, kp + (size_t)(kc + 64) * DD,
                       DD, 64, DD, NN, DD);
      tdm_load_tile_2d((VBUF_OFF + nxt * 64 * 64) * 2, vp + (kc + 64),
                       64, DD, NN, DD, NN);
      __builtin_amdgcn_s_wait_tensorcnt(2);  // current pair landed in LDS
    } else {
      __builtin_amdgcn_s_wait_tensorcnt(0);
    }
    const unsigned short* kt = smem + KBUF_OFF + par * 64 * 64;  // [token][d]
    const unsigned short* vt = smem + VBUF_OFF + par * 64 * 64;  // [d][token]
#else
    const unsigned short* kt = kp + (size_t)kc * DD;
    const unsigned short* vt = vp + kc;
#endif

    // scores for 64 k-tokens: 4 tiles of 16x16
    v8f sc[4];
#pragma unroll
    for (int t = 0; t < 4; ++t) {
#if USE_TDM
      const unsigned short* krow = kt + (t * 16 + col) * DD;
#else
      const unsigned short* krow = kt + (size_t)(t * 16 + col) * DD;
#endif
      v16bf b0 = load_frag(krow, kh * 8, 16 + kh * 8);
      v16bf b1 = load_frag(krow, 32 + kh * 8, 48 + kh * 8);
      v8f z = zero8();
      z = wmma_bf16(qa0, b0, z);
      sc[t] = wmma_bf16(qa1, b1, z);
    }
#pragma unroll
    for (int t = 0; t < 4; ++t)
#pragma unroll
      for (int j = 0; j < 8; ++j) sc[t][j] *= scale;

    // row-wise chunk max (reduce across the 16 lanes of each half)
    float mnew[8], cf[8];
#pragma unroll
    for (int j = 0; j < 8; ++j) {
      float mc = sc[0][j];
#pragma unroll
      for (int t = 1; t < 4; ++t) mc = fmaxf(mc, sc[t][j]);
      mc = fmaxf(mc, __shfl_xor(mc, 1, 32));
      mc = fmaxf(mc, __shfl_xor(mc, 2, 32));
      mc = fmaxf(mc, __shfl_xor(mc, 4, 32));
      mc = fmaxf(mc, __shfl_xor(mc, 8, 32));
      mnew[j] = fmaxf(mrow[j], mc);
      cf[j]   = __expf(mrow[j] - mnew[j]);
      mrow[j] = mnew[j];
    }

    // exponentiate, update running sums, rescale output accumulators
#pragma unroll
    for (int t = 0; t < 4; ++t)
#pragma unroll
      for (int j = 0; j < 8; ++j) sc[t][j] = __expf(sc[t][j] - mnew[j]);
#pragma unroll
    for (int j = 0; j < 8; ++j) {
      float rs = sc[0][j] + sc[1][j] + sc[2][j] + sc[3][j];
      rs += __shfl_xor(rs, 1, 32);
      rs += __shfl_xor(rs, 2, 32);
      rs += __shfl_xor(rs, 4, 32);
      rs += __shfl_xor(rs, 8, 32);
      srow[j] = srow[j] * cf[j] + rs;
    }
#pragma unroll
    for (int t = 0; t < 4; ++t)
#pragma unroll
      for (int j = 0; j < 8; ++j) o[t][j] *= cf[j];

    // probs: C-layout -> LDS [16 rows][64 k] bf16 -> reload in A layout
    __syncthreads();
#pragma unroll
    for (int t = 0; t < 4; ++t)
#pragma unroll
      for (int j = 0; j < 8; ++j) {
        int row = j + 8 * kh;
        smem[PBUF_OFF + row * 64 + t * 16 + col] = f2bf(sc[t][j]);
      }
    __syncthreads();
    v16bf pa0 = load_frag(smem + PBUF_OFF + col * 64, kh * 8, 16 + kh * 8);
    v16bf pa1 = load_frag(smem + PBUF_OFF + col * 64, 32 + kh * 8, 48 + kh * 8);

    // o += P(16x64) * V(64x64); V is [d][n] so B frags are contiguous
#pragma unroll
    for (int dt = 0; dt < 4; ++dt) {
#if USE_TDM
      const unsigned short* vrow = vt + (dt * 16 + col) * 64;
#else
      const unsigned short* vrow = vt + (size_t)(dt * 16 + col) * NN;
#endif
      v16bf vb0 = load_frag(vrow, kh * 8, 16 + kh * 8);
      v16bf vb1 = load_frag(vrow, 32 + kh * 8, 48 + kh * 8);
      o[dt] = wmma_bf16(pa0, vb0, o[dt]);
      o[dt] = wmma_bf16(pa1, vb1, o[dt]);
    }
  }

  // normalize and store attn in [b][n][h*64+d] bf16 for the output GEMM
  const size_t obase = ((size_t)b * NN + qTile * 16) * CC + head * DD;
#pragma unroll
  for (int j = 0; j < 8; ++j) {
    float inv = 1.0f / srow[j];
    int row = j + 8 * kh;
#pragma unroll
    for (int dt = 0; dt < 4; ++dt)
      attn[obase + (size_t)row * CC + dt * 16 + col] = f2bf(o[dt][j] * inv);
  }
}

// ---------------------------------------------------------------------------
// Output projection: [B*N][C] x [C][C]^T -> fp32 d_out. 32x64 tile per wave.
// ---------------------------------------------------------------------------
__global__ __launch_bounds__(32) void out_proj_kernel(
    const unsigned short* __restrict__ attn, const unsigned short* __restrict__ wo,
    float* __restrict__ out) {
  const int lane = threadIdx.x & 31;
  const int col  = lane & 15;
  const int kh   = lane >> 4;

  const int tile  = blockIdx.x;   // 256 m-tiles * 16 n-tiles
  const int mTile = tile & 255;
  const int nTile = tile >> 8;
  const int mBase = mTile * 32;
  const int oBase = nTile * 64;

  v8f acc[2][4];
#pragma unroll
  for (int r = 0; r < 2; ++r)
#pragma unroll
    for (int t = 0; t < 4; ++t) acc[r][t] = zero8();

  const unsigned short* arow0 = attn + (size_t)(mBase + col) * CC;
  const unsigned short* arow1 = attn + (size_t)(mBase + 16 + col) * CC;
  for (int k0 = 0; k0 < CC; k0 += 32) {
    v16bf a0 = load_frag(arow0, k0 + kh * 8, k0 + 16 + kh * 8);
    v16bf a1 = load_frag(arow1, k0 + kh * 8, k0 + 16 + kh * 8);
#pragma unroll
    for (int t = 0; t < 4; ++t) {
      const unsigned short* brow = wo + (size_t)(oBase + t * 16 + col) * CC;
      v16bf b = load_frag(brow, k0 + kh * 8, k0 + 16 + kh * 8);
      acc[0][t] = wmma_bf16(a0, b, acc[0][t]);
      acc[1][t] = wmma_bf16(a1, b, acc[1][t]);
    }
  }

#pragma unroll
  for (int r = 0; r < 2; ++r)
#pragma unroll
    for (int t = 0; t < 4; ++t)
#pragma unroll
      for (int j = 0; j < 8; ++j) {
        int row = r * 16 + j + 8 * kh;
        out[(size_t)(mBase + row) * CC + oBase + t * 16 + col] = acc[r][t][j];
      }
}

// ---------------------------------------------------------------------------
extern "C" void kernel_launch(void* const* d_in, const int* in_sizes, int n_in,
                              void* d_out, int out_size, void* d_ws, size_t ws_size,
                              hipStream_t stream) {
  const float* hidden = (const float*)d_in[0];   // [B,N,C]
  const float* cosb   = (const float*)d_in[1];   // [N,64]
  const float* sinb   = (const float*)d_in[2];   // [N,64]
  const float* qkv_w  = (const float*)d_in[3];   // [3C,C]
  const float* out_w  = (const float*)d_in[4];   // [C,C]
  const float* nqw    = (const float*)d_in[5];   // [64]
  const float* nkw    = (const float*)d_in[6];   // [64]
  float* out = (float*)d_out;

  // Workspace carve-up (bf16 buffers), ~92 MB total
  char* p = (char*)d_ws;
  unsigned short* hb   = (unsigned short*)p; p += (size_t)BB * NN * CC * 2;      // 16.8 MB
  unsigned short* wqkv = (unsigned short*)p; p += (size_t)3 * CC * CC * 2;       //  6.3 MB
  unsigned short* wout = (unsigned short*)p; p += (size_t)CC * CC * 2;           //  2.1 MB
  unsigned short* qbuf = (unsigned short*)p; p += (size_t)BB * HH * NN * DD * 2; // 16.8 MB
  unsigned short* kbuf = (unsigned short*)p; p += (size_t)BB * HH * NN * DD * 2; // 16.8 MB
  unsigned short* vbuf = (unsigned short*)p; p += (size_t)BB * HH * DD * NN * 2; // 16.8 MB
  unsigned short* abuf = (unsigned short*)p; p += (size_t)BB * NN * CC * 2;      // 16.8 MB

  // 1) fp32 -> bf16 casts
  cast_bf16_kernel<<<4096, 256, 0, stream>>>(hidden, hb, BB * NN * CC);
  cast_bf16_kernel<<<2048, 256, 0, stream>>>(qkv_w, wqkv, 3 * CC * CC);
  cast_bf16_kernel<<<1024, 256, 0, stream>>>(out_w, wout, CC * CC);

  // 2) QKV GEMM + RMSNorm + RoPE   (256 m-tiles x 48 n-tiles, 1 wave each)
  qkv_rope_kernel<<<dim3(256 * 48), dim3(32), 0, stream>>>(
      hb, wqkv, cosb, sinb, nqw, nkw, qbuf, kbuf, vbuf);

  // 3) Flash attention             (B*H * N/16 = 8192 waves)
  flash_attn_kernel<<<dim3(BB * HH * (NN / 16)), dim3(32), 0, stream>>>(
      qbuf, kbuf, vbuf, abuf);

  // 4) Output projection           (256 m-tiles x 16 n-tiles)
  out_proj_kernel<<<dim3(256 * 16), dim3(32), 0, stream>>>(abuf, wout, out);
}
